// GraphEncoder_34265249088401
// MI455X (gfx1250) — matrix operand order repaired and checked by reading, not compiled, not run on hardware
//
#include <hip/hip_runtime.h>

// ---------------------------------------------------------------------------
// GGNN GraphEncoder for MI455X (gfx1250, wave32).
//  - All GEMMs: v_wmma_f32_16x16x32_f16, fully unrolled K, A-fragment reuse.
//  - Weights pre-packed to f16 fragment order (32 B per lane per chunk).
//  - GRU gate GEMMs + nonlinearity fused; gates live entirely in LDS.
//  - hc-f16 tile staged to LDS via global_load_async_to_lds_b128 (ASYNCcnt).
//  - agg zeroing fused into the GRU kernel (read-then-zero per element).
// ---------------------------------------------------------------------------

typedef __attribute__((ext_vector_type(16))) _Float16 v16h;
typedef __attribute__((ext_vector_type(8)))  _Float16 v8h;
typedef __attribute__((ext_vector_type(4)))  _Float16 v4h;
typedef __attribute__((ext_vector_type(8)))  float    v8f;
typedef __attribute__((ext_vector_type(4)))  float    v4f;

#define N_NODES 50000
#define N_EDGES 600000
#define IN_DIM  256
#define HID     128
#define NH      (N_NODES * HID)

// -------------------------- CDNA5 async-copy helpers -----------------------
__device__ __forceinline__ unsigned lds_offset(const void* p) {
    // Low 32 bits of a generic LDS address == wave-relative LDS byte offset.
    return (unsigned)(unsigned long long)p;
}
__device__ __forceinline__ void async_copy_b128(unsigned lds, const void* g) {
    asm volatile("global_load_async_to_lds_b128 %0, %1, off"
                 :: "v"(lds), "v"((unsigned long long)g) : "memory");
}
__device__ __forceinline__ void wait_async0() {
    asm volatile("s_wait_asynccnt 0x0" ::: "memory");
}

__device__ __forceinline__ v16h cat8(v8h lo, v8h hi) {
    v16h a;
#pragma unroll
    for (int j = 0; j < 8; ++j) { a[j] = lo[j]; a[j + 8] = hi[j]; }
    return a;
}
__device__ __forceinline__ v8f wmma_f16(v16h a, v16h b, v8f c) {
    return __builtin_amdgcn_wmma_f32_16x16x32_f16(false, a, false, b,
                                                  (short)0, c, false, false);
}

// ---------------------------------------------------------------------------
// Pack B [K,N] f32 (or transpose-source [N,K] if srcT) into f16 fragment
// order: dst[mat][tn16][kc][lane][j], n = tn16*16+(lane&15),
// k = kc*32+(lane>>4)*16+j. One thread per (mat,tn16,kc,lane) -> 32B write.
// ---------------------------------------------------------------------------
__global__ void pack_b_kernel(const float* __restrict__ src, _Float16* __restrict__ dst,
                              int K, int N, int srcT, int nmat)
{
    int idx = blockIdx.x * blockDim.x + threadIdx.x;
    int perMat = (N / 16) * (K / 32) * 32;
    if (idx >= nmat * perMat) return;
    int mat  = idx / perMat;
    int rem  = idx - mat * perMat;
    int lane = rem & 31;
    int rem2 = rem >> 5;
    int kc   = rem2 % (K / 32);
    int tn   = rem2 / (K / 32);
    int half = lane >> 4, l = lane & 15;
    int n = tn * 16 + l;
    const float* s = src + (size_t)mat * K * N;
    _Float16* d = dst + (size_t)mat * K * N + ((size_t)(tn * (K / 32) + kc) * 32 + lane) * 16;
#pragma unroll
    for (int j = 0; j < 16; ++j) {
        int k = kc * 32 + half * 16 + j;
        float v = srcT ? s[(size_t)n * K + k] : s[(size_t)k * N + n];
        d[j] = (_Float16)v;
    }
}

__global__ void f32_to_f16_kernel(const float* __restrict__ in,
                                  _Float16* __restrict__ out, int n)
{
    int idx = blockIdx.x * blockDim.x + threadIdx.x;
    if (idx < n) out[idx] = (_Float16)in[idx];
}

__global__ void zero4_kernel(v4f* __restrict__ p, int n4)
{
    int idx = blockIdx.x * blockDim.x + threadIdx.x;
    if (idx < n4) { v4f z = {0.f, 0.f, 0.f, 0.f}; p[idx] = z; }
}

// ---------------------------------------------------------------------------
// GEMM: C = A[M,K](f16) * Bpack (+bias)(relu?). One wave -> 16x32 tile pair.
// Block = N/32 waves, one 16-row M-tile per block; A staged via async->LDS.
// C (f32) and Ch (f16) outputs both optional.
// ---------------------------------------------------------------------------
template <int K, int NT32>
__global__ void __launch_bounds__(32 * NT32)
gemm_kernel(const _Float16* __restrict__ A, const _Float16* __restrict__ Bp,
            const float* __restrict__ bias, float* __restrict__ C,
            _Float16* __restrict__ Ch, int N, int doRelu)
{
    constexpr int KC = K / 32;
    __shared__ _Float16 At[16 * K];
    const int tid  = threadIdx.x;
    const int wave = tid >> 5, lane = tid & 31;
    const int m0   = blockIdx.x * 16;

    {   // stage 16xK f16 A-tile (contiguous rows) via async b128 copies
        const _Float16* g0 = A + (size_t)m0 * K;
        unsigned l0 = lds_offset(At);
        constexpr int NV = (16 * K) / 8;
#pragma unroll
        for (int i = tid; i < NV; i += 32 * NT32)
            async_copy_b128(l0 + i * 16, g0 + i * 8);
    }
    wait_async0();
    __syncthreads();

    const int half = lane >> 4, l = lane & 15;
    const int tn = wave;
    const _Float16* bp0 = Bp + (size_t)(2 * tn) * KC * 512;
    const _Float16* bp1 = Bp + (size_t)(2 * tn + 1) * KC * 512;

    v8f c0 = {}, c1 = {};
#pragma unroll
    for (int kc = 0; kc < KC; ++kc) {
        v8h alo = *(const v8h*)(At + l * K + kc * 32 + half * 16);
        v8h ahi = *(const v8h*)(At + l * K + kc * 32 + half * 16 + 8);
        v16h a = cat8(alo, ahi);
        v16h b0 = *(const v16h*)(bp0 + (size_t)(kc * 32 + lane) * 16);
        v16h b1 = *(const v16h*)(bp1 + (size_t)(kc * 32 + lane) * 16);
        c0 = wmma_f16(a, b0, c0);
        c1 = wmma_f16(a, b1, c1);
    }

    const int col0 = tn * 32 + l, col1 = col0 + 16;
    const float bv0 = bias ? bias[col0] : 0.0f;
    const float bv1 = bias ? bias[col1] : 0.0f;
#pragma unroll
    for (int r = 0; r < 8; ++r) {
        int row = m0 + half * 8 + r;
        float f0 = c0[r] + bv0, f1 = c1[r] + bv1;
        if (doRelu) { f0 = fmaxf(f0, 0.0f); f1 = fmaxf(f1, 0.0f); }
        if (C) {
            C[(size_t)row * N + col0] = f0;
            C[(size_t)row * N + col1] = f1;
        }
        if (Ch) {
            Ch[(size_t)row * N + col0] = (_Float16)f0;
            Ch[(size_t)row * N + col1] = (_Float16)f1;
        }
    }
}

// ---------------------------------------------------------------------------
// Edge scatter-add: agg[dst[e]] += m16[src[e]] (f16 gather, f32 atomics).
// 32 lanes/edge, 4 contiguous cols/lane (one b64 load per lane).
// ---------------------------------------------------------------------------
__global__ void scatter_add_kernel(const _Float16* __restrict__ m16,
                                   const long long* __restrict__ src,
                                   const long long* __restrict__ dst,
                                   float* __restrict__ agg, int nEdges)
{
    int t = blockIdx.x * blockDim.x + threadIdx.x;
    int e = t >> 5;
    if (e >= nEdges) return;
    int lane = t & 31;
    long long s = src[e];
    long long d = dst[e];
    v4h mv = *(const v4h*)(m16 + (size_t)s * HID + lane * 4);
    float* ap = agg + (size_t)d * HID + lane * 4;
#pragma unroll
    for (int j = 0; j < 4; ++j)
        atomicAdd(ap + j, (float)mv[j]);
}

// ---------------------------------------------------------------------------
// Fused GRU step: gi = agg@Wih^T+bi, gh = hc@Whh^T+bh (WMMA into LDS), then
// hc = GRU(gi, gh, hc). One block = one 16-row M-tile, 8 waves:
//   waves 0-3 -> gi tiles (3 x 32-wide each), waves 4-7 -> gh tiles.
// A fragments loaded once per wave, reused across 24 WMMAs. agg tile is
// converted to f16 in LDS (uniform A path) and the global rows are zeroed
// for the next step's scatter. gi/gh never touch HBM.
// ---------------------------------------------------------------------------
__global__ void __launch_bounds__(256)
gru_fused_kernel(float* __restrict__ agg, const _Float16* __restrict__ hch,
                 const _Float16* __restrict__ wip, const _Float16* __restrict__ whp,
                 const float* __restrict__ bi, const float* __restrict__ bh,
                 float* __restrict__ hc, _Float16* __restrict__ hch_out)
{
    __shared__ float    gl[16 * 384];
    __shared__ float    hl[16 * 384];
    __shared__ _Float16 aggH[16 * 128];
    __shared__ _Float16 hchT[16 * 128];

    const int tid = threadIdx.x, wave = tid >> 5, lane = tid & 31;
    const int m0 = blockIdx.x * 16;

    // async-stage hc-f16 tile (4KB = 256 b128)
    {
        const _Float16* g16 = hch + (size_t)m0 * 128;
        async_copy_b128(lds_offset(hchT) + tid * 16, g16 + tid * 8);
    }
    // convert agg tile to f16 in LDS; zero the global rows behind us
    {
        float* ga = agg + (size_t)m0 * 128;
#pragma unroll
        for (int i = tid; i < 512; i += 256) {
            v4f v = *(const v4f*)(ga + i * 4);
            v4f z = {0.f, 0.f, 0.f, 0.f};
            *(v4f*)(ga + i * 4) = z;
            v4h hv;
#pragma unroll
            for (int j = 0; j < 4; ++j) hv[j] = (_Float16)v[j];
            *(v4h*)(aggH + i * 4) = hv;
        }
    }
    wait_async0();
    __syncthreads();

    const int half = lane >> 4, l = lane & 15;
    const bool isGH = (wave >= 4);                 // wave-uniform
    const _Float16* Asrc = isGH ? hchT : aggH;     // identical layouts
    const _Float16* bp   = isGH ? whp : wip;
    const float*    bias = isGH ? bh : bi;
    float*          outl = isGH ? hl : gl;
    const int tw = wave & 3;                       // covers tiles 3*tw..3*tw+2

    v16h a[4];
#pragma unroll
    for (int kc = 0; kc < 4; ++kc) {
        v8h alo = *(const v8h*)(Asrc + l * 128 + kc * 32 + half * 16);
        v8h ahi = *(const v8h*)(Asrc + l * 128 + kc * 32 + half * 16 + 8);
        a[kc] = cat8(alo, ahi);
    }

    v8f acc[6] = {};
#pragma unroll
    for (int tt = 0; tt < 3; ++tt) {
        int tn = 3 * tw + tt;
#pragma unroll
        for (int kc = 0; kc < 4; ++kc) {
            v16h b0 = *(const v16h*)(bp + ((size_t)(2 * tn) * 4 + kc) * 512 + lane * 16);
            v16h b1 = *(const v16h*)(bp + ((size_t)(2 * tn + 1) * 4 + kc) * 512 + lane * 16);
            acc[2 * tt]     = wmma_f16(a[kc], b0, acc[2 * tt]);
            acc[2 * tt + 1] = wmma_f16(a[kc], b1, acc[2 * tt + 1]);
        }
    }

#pragma unroll
    for (int tt = 0; tt < 3; ++tt) {
        int col0 = (3 * tw + tt) * 32 + l, col1 = col0 + 16;
        float bv0 = bias[col0], bv1 = bias[col1];
#pragma unroll
        for (int r = 0; r < 8; ++r) {
            int row = half * 8 + r;
            outl[row * 384 + col0] = acc[2 * tt][r] + bv0;
            outl[row * 384 + col1] = acc[2 * tt + 1][r] + bv1;
        }
    }
    __syncthreads();

    // GRU nonlinearity: 16x128 elements, 8 per thread.
#pragma unroll
    for (int e = tid; e < 2048; e += 256) {
        int r = e >> 7, c = e & 127;
        float ir = gl[r * 384 + c], iz = gl[r * 384 + 128 + c], in = gl[r * 384 + 256 + c];
        float hr = hl[r * 384 + c], hz = hl[r * 384 + 128 + c], hn = hl[r * 384 + 256 + c];
        float rr = 1.0f / (1.0f + __expf(-(ir + hr)));
        float zz = 1.0f / (1.0f + __expf(-(iz + hz)));
        float nn = tanhf(in + rr * hn);
        size_t gidx = (size_t)(m0 + r) * 128 + c;
        float hv = hc[gidx];
        float nh = (1.0f - zz) * nn + zz * hv;
        hc[gidx] = nh;
        hch_out[gidx] = (_Float16)nh;
    }
}

// ---------------------------------------------------------------------------
// LayerNorm(H=128) + optional residual + relu; also refreshes hc / hc_f16.
// ---------------------------------------------------------------------------
__global__ void __launch_bounds__(128)
layernorm_kernel(const float* __restrict__ hcin, const float* __restrict__ resid,
                 const float* __restrict__ g, const float* __restrict__ b,
                 float* __restrict__ out, float* __restrict__ hc_out,
                 _Float16* __restrict__ hch_out, int nRows, int useResid)
{
    int wave = threadIdx.x >> 5, lane = threadIdx.x & 31;
    int row = blockIdx.x * 4 + wave;
    if (row >= nRows) return;
    const float* hp = hcin + (size_t)row * HID;

    float v[4], s = 0.0f;
#pragma unroll
    for (int j = 0; j < 4; ++j) { v[j] = hp[lane + 32 * j]; s += v[j]; }
#pragma unroll
    for (int off = 16; off > 0; off >>= 1) s += __shfl_xor(s, off, 32);
    float mu = s * (1.0f / 128.0f);

    float q = 0.0f;
#pragma unroll
    for (int j = 0; j < 4; ++j) { float d = v[j] - mu; q += d * d; }
#pragma unroll
    for (int off = 16; off > 0; off >>= 1) q += __shfl_xor(q, off, 32);
    float inv = rsqrtf(q * (1.0f / 128.0f) + 1e-5f);

#pragma unroll
    for (int j = 0; j < 4; ++j) {
        int c = lane + 32 * j;
        float o = (v[j] - mu) * inv * g[c] + b[c];
        if (useResid) o += resid[(size_t)row * HID + c];
        o = fmaxf(o, 0.0f);
        out[(size_t)row * HID + c] = o;
        if (hc_out)  hc_out[(size_t)row * HID + c] = o;
        if (hch_out) hch_out[(size_t)row * HID + c] = (_Float16)o;
    }
}

// ---------------------------------------------------------------------------
extern "C" void kernel_launch(void* const* d_in, const int* in_sizes, int n_in,
                              void* d_out, int out_size, void* d_ws, size_t ws_size,
                              hipStream_t stream)
{
    const float*     x      = (const float*)d_in[0];
    const long long* ei     = (const long long*)d_in[1];   // int64 [2, E]
    const float*     in_w   = (const float*)d_in[2];       // [256,128]
    const float*     in_b   = (const float*)d_in[3];
    const float*     ggnn_w = (const float*)d_in[4];       // [3,5,128,128]
    const float*     w_ih   = (const float*)d_in[5];       // [3,384,128]
    const float*     w_hh   = (const float*)d_in[6];       // [3,384,128]
    const float*     b_ih   = (const float*)d_in[7];       // [3,384]
    const float*     b_hh   = (const float*)d_in[8];       // [3,384]
    const float*     ln_g   = (const float*)d_in[9];       // [3,128]
    const float*     ln_b   = (const float*)d_in[10];      // [3,128]
    float* out = (float*)d_out;

    float* ws  = (float*)d_ws;
    float* hc  = ws;            // [N,H] f32
    float* h   = hc  + NH;      // [N,H] f32 (layer input / residual)
    float* agg = h   + NH;      // [N,H] f32
    _Float16* m16   = (_Float16*)(agg + NH);            // [N,H] f16 (messages)
    _Float16* hch   = m16 + NH;                         // [N,H] f16 (hc shadow)
    _Float16* xh    = hch + NH;                         // [N,256] f16
    _Float16* inwP  = xh + (size_t)N_NODES * IN_DIM;    // packed 256x128
    _Float16* ggnnP = inwP + IN_DIM * HID;              // packed 15x 128x128
    _Float16* wihP  = ggnnP + 15 * HID * HID;           // packed 3x 128x384
    _Float16* whhP  = wihP + 3 * HID * 384;             // packed 3x 128x384

    const long long* src = ei;
    const long long* dst = ei + N_EDGES;

    // ---- one-time packs / converts / poison-proof agg zero -------------
    pack_b_kernel<<<(2048 + 255) / 256, 256, 0, stream>>>(in_w, inwP, IN_DIM, HID, 0, 1);
    pack_b_kernel<<<(15 * 1024 + 255) / 256, 256, 0, stream>>>(ggnn_w, ggnnP, HID, HID, 0, 15);
    pack_b_kernel<<<(3 * 3072 + 255) / 256, 256, 0, stream>>>(w_ih, wihP, HID, 384, 1, 3);
    pack_b_kernel<<<(3 * 3072 + 255) / 256, 256, 0, stream>>>(w_hh, whhP, HID, 384, 1, 3);
    f32_to_f16_kernel<<<(N_NODES * IN_DIM + 255) / 256, 256, 0, stream>>>(
        x, xh, N_NODES * IN_DIM);
    zero4_kernel<<<(NH / 4 + 255) / 256, 256, 0, stream>>>((v4f*)agg, NH / 4);

    // ---- input projection: hc = relu(x @ in_w + in_b) ------------------
    gemm_kernel<IN_DIM, 4><<<N_NODES / 16, 128, 0, stream>>>(
        xh, inwP, in_b, hc, hch, HID, 1);

    for (int i = 0; i < 3; ++i) {
        for (int s = 0; s < 5; ++s) {
            gemm_kernel<HID, 4><<<N_NODES / 16, 128, 0, stream>>>(
                hch, ggnnP + (size_t)(i * 5 + s) * HID * HID, nullptr,
                nullptr, m16, HID, 0);

            scatter_add_kernel<<<(N_EDGES * 32) / 256, 256, 0, stream>>>(
                m16, src, dst, agg, N_EDGES);

            gru_fused_kernel<<<N_NODES / 16, 256, 0, stream>>>(
                agg, hch, wihP + (size_t)i * HID * 384, whhP + (size_t)i * HID * 384,
                b_ih + (size_t)i * 384, b_hh + (size_t)i * 384, hc, hch);
        }
        float* outp = (i == 2) ? out : h;
        layernorm_kernel<<<(N_NODES + 3) / 4, 128, 0, stream>>>(
            hc, h, ln_g + (size_t)i * HID, ln_b + (size_t)i * HID, outp,
            (i < 2) ? hc : nullptr, (i < 2) ? hch : nullptr, N_NODES, i > 0 ? 1 : 0);
    }
}
